// Pgot_70798240907875
// MI455X (gfx1250) — matrix operand order repaired and checked by reading, not compiled, not run on hardware
//
#include <hip/hip_runtime.h>
#include <hip/hip_bf16.h>

typedef __attribute__((ext_vector_type(16))) _Float16 v16h;
typedef __attribute__((ext_vector_type(8)))  _Float16 v8h;
typedef __attribute__((ext_vector_type(4)))  _Float16 v4h;
typedef __attribute__((ext_vector_type(8)))  float    v8f;

// ---------------- device helpers ----------------

__device__ __forceinline__ v16h make_v16(v8h lo, v8h hi) {
  return __builtin_shufflevector(lo, hi, 0,1,2,3,4,5,6,7,8,9,10,11,12,13,14,15);
}
// A-fragment (16x32 f16): lane<16 holds K = kb..kb+7 and kb+16..kb+23 (kb = 8*(lane>>4))
__device__ __forceinline__ v16h ldfragA(const _Float16* row, int kbase) {
  v8h lo = *(const v8h*)(row + kbase);
  v8h hi = *(const v8h*)(row + kbase + 16);
  return make_v16(lo, hi);
}
// B-fragment from B^T layout [n][k]: lane<16 holds K = kb..kb+7 and kb+16..kb+23 (kb = 8*(lane>>4))
__device__ __forceinline__ v16h ldfragB(const _Float16* row, int koff) {
  v8h lo = *(const v8h*)(row + koff);
  v8h hi = *(const v8h*)(row + koff + 16);
  return make_v16(lo, hi);
}
__device__ __forceinline__ v8h cvt8(float4 a, float4 b) {
  v8h h;
  h[0] = (_Float16)a.x; h[1] = (_Float16)a.y; h[2] = (_Float16)a.z; h[3] = (_Float16)a.w;
  h[4] = (_Float16)b.x; h[5] = (_Float16)b.y; h[6] = (_Float16)b.z; h[7] = (_Float16)b.w;
  return h;
}
__device__ __forceinline__ v4h cvt4(float4 a) {
  v4h h;
  h[0] = (_Float16)a.x; h[1] = (_Float16)a.y; h[2] = (_Float16)a.z; h[3] = (_Float16)a.w;
  return h;
}
__device__ __forceinline__ float gelu_f(float x) {
  float x3 = x * x * x;
  return 0.5f * x * (1.0f + tanhf(0.7978845608028654f * (x + 0.044715f * x3)));
}

// ---------------- generic batched WMMA GEMM ----------------
// C[bz][m][n] = sum_k A[bz][m][k] * B[bz][k][n] (+bias) (+act)
// EXACT-TILE kernel: requires M%64==0, N%64==0, K%32==0 (all call sites satisfy this).
// 64x64 tile per block, 128 threads = 4 waves, each wave 2x2 WMMA 16x16x32 tiles.
__global__ void __launch_bounds__(128)
k_gemm(const float* __restrict__ A, long aBS, int lda,
       const float* __restrict__ B, long bBS, int ldb,
       const float* __restrict__ bias, int biasPerRow,
       float* __restrict__ C, long cBS, int ldc,
       int M, int N, int K, int act) {
  __shared__ alignas(16) _Float16 As[64][40];
  __shared__ alignas(16) _Float16 Bs[64][40]; // stores B^T tile: Bs[n][k]
  const int tid = threadIdx.x;
  const float* Ab = A + (long)blockIdx.z * aBS;
  const float* Bb = B + (long)blockIdx.z * bBS;
  float* Cb = C + (long)blockIdx.z * cBS;
  const int m0 = blockIdx.y * 64, n0 = blockIdx.x * 64;
  const int wave = tid >> 5, lane = tid & 31;
  const int wm = (wave >> 1) * 32, wn = (wave & 1) * 32;
  const int mlo = lane & 15, hi = lane >> 4, kbase = hi * 8;
  // A tile mapping: row = tid>>1, 16-col half = (tid&1)*16  (coalesced b128 loads)
  const int ra = tid >> 1, ca = (tid & 1) * 16;
  // B tile mapping: k-row = tid>>2, 16-col group = (tid&3)*16 (coalesced b128 loads)
  const int kb = tid >> 2, nb = (tid & 3) * 16;
  v8f acc[2][2] = {};
  for (int k0 = 0; k0 < K; k0 += 32) {
    const float* ap = Ab + (long)(m0 + ra) * lda + (k0 + ca);
    float4 a0 = *(const float4*)(ap + 0);
    float4 a1 = *(const float4*)(ap + 4);
    float4 a2 = *(const float4*)(ap + 8);
    float4 a3 = *(const float4*)(ap + 12);
    const float* bp = Bb + (long)(k0 + kb) * ldb + (n0 + nb);
    float4 b0 = *(const float4*)(bp + 0);
    float4 b1 = *(const float4*)(bp + 4);
    float4 b2 = *(const float4*)(bp + 8);
    float4 b3 = *(const float4*)(bp + 12);
    if (k0 + 32 < K) { // uniform branch: prefetch next K tile (global_prefetch_b8)
      __builtin_prefetch(ap + 32, 0, 3);
      __builtin_prefetch(bp + (long)32 * ldb, 0, 3);
    }
    *(v8h*)&As[ra][ca] = cvt8(a0, a1);
    *(v8h*)&As[ra][ca + 8] = cvt8(a2, a3);
    Bs[nb +  0][kb] = (_Float16)b0.x; Bs[nb +  1][kb] = (_Float16)b0.y;
    Bs[nb +  2][kb] = (_Float16)b0.z; Bs[nb +  3][kb] = (_Float16)b0.w;
    Bs[nb +  4][kb] = (_Float16)b1.x; Bs[nb +  5][kb] = (_Float16)b1.y;
    Bs[nb +  6][kb] = (_Float16)b1.z; Bs[nb +  7][kb] = (_Float16)b1.w;
    Bs[nb +  8][kb] = (_Float16)b2.x; Bs[nb +  9][kb] = (_Float16)b2.y;
    Bs[nb + 10][kb] = (_Float16)b2.z; Bs[nb + 11][kb] = (_Float16)b2.w;
    Bs[nb + 12][kb] = (_Float16)b3.x; Bs[nb + 13][kb] = (_Float16)b3.y;
    Bs[nb + 14][kb] = (_Float16)b3.z; Bs[nb + 15][kb] = (_Float16)b3.w;
    __syncthreads();
    v16h af[2], bf[2];
    for (int i = 0; i < 2; i++) af[i] = ldfragA(&As[wm + i * 16 + mlo][0], kbase);
    for (int j = 0; j < 2; j++) bf[j] = ldfragB(&Bs[wn + j * 16 + mlo][0], kbase);
    for (int i = 0; i < 2; i++)
      for (int j = 0; j < 2; j++)
        acc[i][j] = __builtin_amdgcn_wmma_f32_16x16x32_f16(
            false, af[i], false, bf[j], (short)0, acc[i][j], false, false);
    __syncthreads();
  }
  for (int i = 0; i < 2; i++)
    for (int j = 0; j < 2; j++)
      for (int r = 0; r < 8; r++) {
        int gm = m0 + wm + i * 16 + hi * 8 + r;
        int gn = n0 + wn + j * 16 + mlo;
        float v = acc[i][j][r];
        if (bias) v += biasPerRow ? bias[gm] : bias[gn];
        if (act == 1) v = v > 0.f ? v : 0.f;
        else if (act == 2) v = gelu_f(v);
        Cb[(long)gm * ldc + gn] = v;
      }
}

// ---------------- fused attention (one block per bt) ----------------
// Q,K,V,O: [768][128][128] f32, bias: [768][128][128]
__global__ void __launch_bounds__(128)
k_attention(const float* __restrict__ Q, const float* __restrict__ Kg,
            const float* __restrict__ V, const float* __restrict__ bias,
            float* __restrict__ O) {
  __shared__ alignas(16) _Float16 Qh[128][40];   // K padded 16->32 with zeros
  __shared__ alignas(16) _Float16 Kh[128][40];
  __shared__ alignas(16) _Float16 Sc[128][136];  // scores -> probs (f16)
  __shared__ alignas(16) _Float16 Vt[16][136];   // V^T per head: [d][k]
  const int bt = blockIdx.x, tid = threadIdx.x;
  const int wave = tid >> 5, lane = tid & 31;
  const int mlo = lane & 15, hi = lane >> 4, kbase = hi * 8;
  const long base = (long)bt * 16384;
  // zero the K-pad region (cols 16..31) once; head loads only touch cols 0..15
  {
    v8h z = {};
    for (int idx = tid; idx < 256; idx += 128) {
      int s = idx >> 1, off = 16 + (idx & 1) * 8;
      *(v8h*)&Qh[s][off] = z;
      *(v8h*)&Kh[s][off] = z;
    }
  }
  __syncthreads();
  for (int h = 0; h < 8; h++) {
    for (int idx = tid; idx < 512; idx += 128) {
      int s = idx >> 2, c = (idx & 3) * 4;
      float4 qv = *(const float4*)(Q  + base + (long)s * 128 + h * 16 + c);
      float4 kv = *(const float4*)(Kg + base + (long)s * 128 + h * 16 + c);
      *(v4h*)&Qh[s][c] = cvt4(qv);
      *(v4h*)&Kh[s][c] = cvt4(kv);
    }
    for (int idx = tid; idx < 512; idx += 128) {
      int kk = idx >> 2, dq = (idx & 3) * 4;
      float4 vv = *(const float4*)(V + base + (long)kk * 128 + h * 16 + dq);
      Vt[dq + 0][kk] = (_Float16)vv.x;
      Vt[dq + 1][kk] = (_Float16)vv.y;
      Vt[dq + 2][kk] = (_Float16)vv.z;
      Vt[dq + 3][kk] = (_Float16)vv.w;
    }
    __syncthreads();
    // scores = Q K^T / 4 + bias
    for (int mi = 0; mi < 2; mi++) {
      int mt = wave * 2 + mi;
      v16h af = ldfragA(&Qh[mt * 16 + mlo][0], kbase);
      for (int nt = 0; nt < 8; nt++) {
        v16h bf = ldfragB(&Kh[nt * 16 + mlo][0], kbase);
        v8f acc = {};
        acc = __builtin_amdgcn_wmma_f32_16x16x32_f16(
            false, af, false, bf, (short)0, acc, false, false);
        for (int r = 0; r < 8; r++) {
          int gm = mt * 16 + hi * 8 + r;
          int gn = nt * 16 + mlo;
          float sv = acc[r] * 0.25f + bias[base + (long)gm * 128 + gn];
          Sc[gm][gn] = (_Float16)sv;
        }
      }
    }
    __syncthreads();
    // row softmax (one thread per row)
    {
      int r = tid;
      float mx = -1e30f;
      for (int n = 0; n < 128; n++) { float v = (float)Sc[r][n]; mx = fmaxf(mx, v); }
      float sum = 0.f;
      for (int n = 0; n < 128; n++) {
        float e = expf((float)Sc[r][n] - mx);
        sum += e;
        Sc[r][n] = (_Float16)e;
      }
      float inv = 1.f / sum;
      for (int n = 0; n < 128; n++) Sc[r][n] = (_Float16)((float)Sc[r][n] * inv);
    }
    __syncthreads();
    // O[:, h*16:(h+1)*16] = attn @ V_h   (K = 128 over 4 WMMA steps)
    for (int mi = 0; mi < 2; mi++) {
      int mt = wave * 2 + mi;
      v8f acc = {};
      for (int ks = 0; ks < 4; ks++) {
        v16h af = ldfragA(&Sc[mt * 16 + mlo][0], ks * 32 + kbase);
        v16h bf = ldfragB(&Vt[mlo][0], ks * 32 + kbase);
        acc = __builtin_amdgcn_wmma_f32_16x16x32_f16(
            false, af, false, bf, (short)0, acc, false, false);
      }
      for (int r = 0; r < 8; r++) {
        int gm = mt * 16 + hi * 8 + r;
        O[base + (long)gm * 128 + h * 16 + mlo] = acc[r];
      }
    }
    __syncthreads();
  }
}

// ---------------- small VALU kernels ----------------

__global__ void k_tmark(const float* xm, const float* W, const float* b, float* out) {
  int bt = blockIdx.x, d = threadIdx.x;
  float s = b[d];
  for (int j = 0; j < 4; j++) s += xm[bt * 4 + j] * W[j * 128 + d];
  out[bt * 128 + d] = s;
}

__global__ void k_smark(const float* sx, const float* W, const float* b, float* out) {
  int s = blockIdx.x, d = threadIdx.x;
  out[s * 128 + d] = b[d] + sx[s * 2] * W[d] + sx[s * 2 + 1] * W[128 + d];
}

__global__ void k_embed(const float* xe, const float* vW, const float* vb,
                        const float* tm, const float* sm, float* H) {
  int bt = blockIdx.x, d = threadIdx.x;
  float w = vW[d], b0 = vb[d], t = tm[bt * 128 + d];
  long base = (long)bt * 16384;
  for (int s = 0; s < 128; s++)
    H[base + (long)s * 128 + d] = xe[bt * 128 + s] * w + b0 + t + sm[s * 128 + d];
}

__device__ __forceinline__ float kmat(int i, int j) {
  int ri = i >> 3, ci = i & 7, rj = j >> 3, cj = j & 7;
  int dr = ri - rj; if (dr < 0) dr = -dr;
  int dc = ci - cj; if (dc < 0) dc = -dc;
  return (dr <= 1 && dc <= 1 && (dr | dc)) ? 0.135335283f /*e^-2*/ : 0.018315639f /*e^-4*/;
}

__global__ void k_sinkhorn(const float* __restrict__ H, float* __restrict__ bias) {
  __shared__ float a[128], u[128], v[128], red[128];
  int bt = blockIdx.x, i = threadIdx.x;
  const float* Hr = H + ((long)bt * 128 + i) * 128;
  float p = 0.f;
  for (int d = 0; d < 128; d++) p += Hr[d];
  p *= (1.f / 128.f);
  red[i] = p; __syncthreads();
  for (int off = 64; off > 0; off >>= 1) { if (i < off) red[i] = fmaxf(red[i], red[i + off]); __syncthreads(); }
  float mx = red[0]; __syncthreads();
  float e = expf(p - mx);
  red[i] = e; __syncthreads();
  for (int off = 64; off > 0; off >>= 1) { if (i < off) red[i] += red[i + off]; __syncthreads(); }
  a[i] = e / red[0];
  v[i] = 1.f / 128.f;
  __syncthreads();
  for (int it = 0; it < 5; it++) {
    float s = 0.f;
    for (int j = 0; j < 128; j++) s += kmat(i, j) * v[j];
    u[i] = a[i] / (s + 1e-8f);
    __syncthreads();
    s = 0.f;
    for (int j = 0; j < 128; j++) s += kmat(j, i) * u[j];
    v[i] = a[i] / (s + 1e-8f);
    __syncthreads();
  }
  float ui = u[i];
  float* br = bias + ((long)bt * 128 + i) * 128;
  for (int j = 0; j < 128; j++) br[j] = ui * kmat(i, j) * v[j];
}

// out = LN(x + y) (block per row, 128 threads)
__global__ void k_add_ln(const float* x, const float* y,
                         const float* g, const float* b, float* outp) {
  __shared__ float red[128];
  long row = blockIdx.x;
  int d = threadIdx.x;
  float v = x[row * 128 + d] + y[row * 128 + d];
  red[d] = v; __syncthreads();
  for (int off = 64; off > 0; off >>= 1) { if (d < off) red[d] += red[d + off]; __syncthreads(); }
  float m = red[0] * (1.f / 128.f); __syncthreads();
  float c = v - m;
  red[d] = c * c; __syncthreads();
  for (int off = 64; off > 0; off >>= 1) { if (d < off) red[d] += red[d + off]; __syncthreads(); }
  float var = red[0] * (1.f / 128.f);
  outp[row * 128 + d] = c * rsqrtf(var + 1e-5f) * g[d] + b[d];
}

// H_local[bt][d][s] = x[bt][s][d]
__global__ void k_transpose_bt(const float* x, float* o) {
  int bt = blockIdx.x, d = threadIdx.x;
  long base = (long)bt * 16384;
  for (int s = 0; s < 128; s++) o[base + (long)d * 128 + s] = x[base + (long)s * 128 + d];
}

// [B,T,S,D] -> [B,S,T,D]
__global__ void k_gather_ts(const float* in, float* out) {
  long r = blockIdx.x; int d = threadIdx.x;
  int t = (int)(r % 96); long bs = r / 96; int s = (int)(bs % 128); int b = (int)(bs / 128);
  long inRow = ((long)b * 96 + t) * 128 + s;
  out[r * 128 + d] = in[inRow * 128 + d];
}

// [B,S,T,D] -> [B,T,S,D]
__global__ void k_scatter_ts(const float* in, float* out) {
  long r = blockIdx.x; int d = threadIdx.x;
  int s = (int)(r % 128); long btl = r / 128; int t = (int)(btl % 96); int b = (int)(btl / 96);
  long inRow = ((long)b * 128 + s) * 96 + t;
  out[r * 128 + d] = in[inRow * 128 + d];
}

// gated fusion: pooled->gate->Hf
__global__ void k_gate(const float* HL, const float* HG,
                       const float* Wg, const float* bg, float* Hf) {
  __shared__ float pooled[128], g[128];
  int bt = blockIdx.x, i = threadIdx.x;
  long base = (long)bt * 16384;
  float s = 0.f;
  for (int j = 0; j < 128; j++) s += HL[base + (long)i * 128 + j] + HG[base + (long)i * 128 + j];
  pooled[i] = s * (1.f / 128.f);
  __syncthreads();
  float acc = bg[i];
  for (int d = 0; d < 128; d++) acc += pooled[d] * Wg[i * 128 + d]; // pooled @ Wg.T
  g[i] = 1.f / (1.f + expf(-acc));
  __syncthreads();
  float gi = g[i];
  for (int j = 0; j < 128; j++) {
    long idx = base + (long)i * 128 + j;
    Hf[idx] = gi * HL[idx] + (1.f - gi) * HG[idx];
  }
}

// Yt[b][j][i] = HR[b*96 + i/128][i%128][j]
__global__ void k_build_yt(const float* HR, float* Yt) {
  long blk = blockIdx.x; int j = threadIdx.x;
  int i = (int)(blk % 12288); int b = (int)(blk / 12288);
  int t = i >> 7, e = i & 127;
  Yt[((long)b * 128 + j) * 12288 + i] = HR[(((long)b * 96 + t) * 128 + e) * 128 + j];
}

// out[b][o][s2] = sum_j Yw[b][j][o*128+s2] * Wr[j] + br  (relu already applied)
__global__ void k_head(const float* Yw, const float* Wr, const float* br, float* out) {
  int blk = blockIdx.x; int s2 = threadIdx.x;
  int o = blk % 24; int b = blk / 24;
  float acc = 0.f;
  for (int j = 0; j < 128; j++)
    acc += Yw[((long)b * 128 + j) * 3072 + o * 128 + s2] * Wr[j];
  out[((long)b * 24 + o) * 128 + s2] = acc + br[0];
}

// ---------------- host driver ----------------

extern "C" void kernel_launch(void* const* d_in, const int* in_sizes, int n_in,
                              void* d_out, int out_size, void* d_ws, size_t ws_size,
                              hipStream_t stream) {
  (void)in_sizes; (void)n_in; (void)out_size; (void)ws_size;
  const float* x_enc = (const float*)d_in[0];
  const float* x_mark = (const float*)d_in[1];
  const float* smx  = (const float*)d_in[2];
  const float* val_W = (const float*)d_in[3];
  const float* val_b = (const float*)d_in[4];
  const float* tmW  = (const float*)d_in[5];
  const float* tmb  = (const float*)d_in[6];
  const float* smW  = (const float*)d_in[7];
  const float* smb  = (const float*)d_in[8];
  const float* Wq   = (const float*)d_in[9];
  const float* Wk   = (const float*)d_in[10];
  const float* Wv   = (const float*)d_in[11];
  const float* Wo   = (const float*)d_in[12];
  const float* ln1g = (const float*)d_in[13];
  const float* ln1b = (const float*)d_in[14];
  const float* W1   = (const float*)d_in[15];
  const float* b1   = (const float*)d_in[16];
  const float* W2   = (const float*)d_in[17];
  const float* b2   = (const float*)d_in[18];
  const float* ln2g = (const float*)d_in[19];
  const float* ln2b = (const float*)d_in[20];
  const float* Ws1  = (const float*)d_in[21];
  const float* bs1  = (const float*)d_in[22];
  const float* Ws2  = (const float*)d_in[23];
  const float* bs2  = (const float*)d_in[24];
  const float* Wt1  = (const float*)d_in[25];
  const float* bt1  = (const float*)d_in[26];
  const float* Wt2  = (const float*)d_in[27];
  const float* bt2  = (const float*)d_in[28];
  const float* Wp   = (const float*)d_in[29];
  const float* bp   = (const float*)d_in[30];
  const float* Wsp  = (const float*)d_in[31];
  const float* bsp  = (const float*)d_in[32];
  const float* Wc   = (const float*)d_in[33];
  const float* bc   = (const float*)d_in[34];
  const float* Wg   = (const float*)d_in[35];
  const float* bg   = (const float*)d_in[36];
  const float* Wr   = (const float*)d_in[37];
  const float* br   = (const float*)d_in[38];

  const long NBT = 768L * 128 * 128; // 12,582,912 floats
  float* ws   = (float*)d_ws;
  float* tmark = ws;                 // 98304
  float* smark = tmark + 98304;      // 16384
  float* H    = smark + 16384;
  float* biasb = H + NBT;
  float* x    = biasb + NBT;
  float* bq   = x + NBT;
  float* bk   = bq + NBT;
  float* bv   = bk + NBT;
  float* t1   = bv + NBT;
  float* t2   = t1 + NBT;
  float* yw   = t2 + NBT;            // 8*128*3072 = 3,145,728

  auto gemm = [&](const float* A, long aBS, int lda, const float* Bm, long bBS, int ldb,
                  const float* bias, int bpr, float* C, long cBS, int ldc,
                  int M, int N, int K, int act, int batch) {
    dim3 grid((N + 63) / 64, (M + 63) / 64, batch);
    k_gemm<<<grid, dim3(128), 0, stream>>>(A, aBS, lda, Bm, bBS, ldb, bias, bpr,
                                           C, cBS, ldc, M, N, K, act);
  };

  // ---- embedding + OT bias ----
  k_tmark<<<768, 128, 0, stream>>>(x_mark, tmW, tmb, tmark);
  k_smark<<<128, 128, 0, stream>>>(smx, smW, smb, smark);
  k_embed<<<768, 128, 0, stream>>>(x_enc, val_W, val_b, tmark, smark, H);
  k_sinkhorn<<<768, 128, 0, stream>>>(H, biasb);
  hipMemcpyAsync(x, H, sizeof(float) * NBT, hipMemcpyDeviceToDevice, stream);

  // ---- 3 encoder layers ----
  for (int li = 0; li < 3; li++) {
    const float* Wq_l = Wq + (long)li * 16384;
    const float* Wk_l = Wk + (long)li * 16384;
    const float* Wv_l = Wv + (long)li * 16384;
    const float* Wo_l = Wo + (long)li * 16384;
    gemm(x, 0, 128, Wq_l, 0, 128, nullptr, 0, bq, 0, 128, 98304, 128, 128, 0, 1);
    gemm(x, 0, 128, Wk_l, 0, 128, nullptr, 0, bk, 0, 128, 98304, 128, 128, 0, 1);
    gemm(x, 0, 128, Wv_l, 0, 128, nullptr, 0, bv, 0, 128, 98304, 128, 128, 0, 1);
    k_attention<<<768, 128, 0, stream>>>(bq, bk, bv, biasb, t2);
    gemm(t2, 0, 128, Wo_l, 0, 128, nullptr, 0, t1, 0, 128, 98304, 128, 128, 0, 1);
    k_add_ln<<<98304, 128, 0, stream>>>(x, t1, ln1g + li * 128, ln1b + li * 128, x);
    // FFN (chunked so the [rows x 512] scratch fits in bq)
    const int CH = 16384;
    for (int c = 0; c < 6; c++) {
      const float* xc = x + (long)c * CH * 128;
      gemm(xc, 0, 128, W1 + (long)li * 128 * 512, 0, 512, b1 + li * 512, 0,
           bq, 0, 512, CH, 512, 128, 2 /*gelu*/, 1);
      gemm(bq, 0, 512, W2 + (long)li * 512 * 128, 0, 128, b2 + li * 128, 0,
           t1 + (long)c * CH * 128, 0, 128, CH, 128, 512, 0, 1);
    }
    k_add_ln<<<98304, 128, 0, stream>>>(x, t1, ln2g + li * 128, ln2b + li * 128, x);
  }

  // ---- H_local ----
  k_transpose_bt<<<768, 128, 0, stream>>>(x, bv); // HL in bv

  // ---- Hs = relu((H @ Ws1 + bs1) @ Ws2 + bs2)  (chunked) ----
  for (int c = 0; c < 2; c++) {
    long off = (long)c * 49152;
    gemm(H + off * 128, 0, 128, Ws1, 0, 256, bs1, 0, bq, 0, 256, 49152, 256, 128, 0, 1);
    gemm(bq, 0, 256, Ws2, 0, 128, bs2, 0, t1 + off * 128, 0, 128, 49152, 128, 256, 1, 1);
  }
  // ---- temporal mixing ----
  k_gather_ts<<<98304, 128, 0, stream>>>(t1, t2); // [B,S,T,D]
  for (int c = 0; c < 2; c++) {
    long off = (long)c * 49152;
    gemm(t2 + off * 128, 0, 128, Wt1, 0, 256, bt1, 0, bq, 0, 256, 49152, 256, 128, 0, 1);
    gemm(bq, 0, 256, Wt2, 0, 128, bt2, 0, t1 + off * 128, 0, 128, 49152, 128, 256, 1, 1);
  }
  k_scatter_ts<<<98304, 128, 0, stream>>>(t1, t2); // back to [B,T,S,D]
  // ---- H_global = Htb @ Wp + bp ----
  gemm(t2, 0, 128, Wp, 0, 128, bp, 0, t1, 0, 128, 98304, 128, 128, 0, 1); // HG in t1

  // ---- gated fusion ----
  k_gate<<<768, 128, 0, stream>>>(bv, t1, Wg, bg, t2); // Hf in t2
  // ---- 1x1 conv refine: HR[bt] = Wc @ Hf[bt] + bc  (A shared, B per-batch, bias per-row)
  gemm(Wc, 0, 128, t2, 16384, 128, bc, 1, t1, 16384, 128, 128, 128, 128, 0, 768); // HR in t1

  // ---- horizon projection ----
  k_build_yt<<<98304, 128, 0, stream>>>(t1, t2); // Yt [8][128][12288]
  gemm(t2, 128L * 12288, 12288, Wsp, 0, 3072, bsp, 0,
       yw, 128L * 3072, 3072, 128, 3072, 12288, 1 /*relu*/, 8);
  // ---- regression head ----
  k_head<<<192, 128, 0, stream>>>(yw, Wr, br, (float*)d_out);
}
